// NonLocalAttention_29746943492347
// MI455X (gfx1250) — matrix operand order repaired and checked by reading, not compiled, not run on hardware
//
#include <hip/hip_runtime.h>
#include <hip/hip_bf16.h>

typedef __attribute__((ext_vector_type(16))) _Float16 v16h;
typedef __attribute__((ext_vector_type(8)))  float    v8f;

#define NHEADS 4
#define ED     8
#define DIM    32
#define PS     7
#define PS2    49
#define PAD_   3
#define KTOP   10
#define T_     3
#define H_     32
#define W_     32
#define LQ     3072          // T*H*W
#define PDIM   392           // ED*PS*PS
#define KPAD   416           // 13 * 32
#define KBLK   13
#define QTILES 192           // LQ / 16
#define CHUNK  1536
#define NCHUNK 2
#define CTPC   96            // CHUNK / 16

// ---------------- patch element fetch (shared helper) ----------------
__device__ __forceinline__ float patch_fetch(const float* __restrict__ feat,
                                             int h, int q, int pd) {
  int ed = pd / PS2, r = pd % PS2;
  int di = r / PS, dj = r % PS;
  int t = q >> 10, rem = q & 1023;
  int y = rem >> 5, x = rem & 31;
  int yy = y + di - PAD_, xx = x + dj - PAD_;
  if ((unsigned)yy >= (unsigned)H_ || (unsigned)xx >= (unsigned)W_) return 0.0f;
  int c = h * ED + ed;
  return feat[((t * DIM + c) * H_ + yy) * W_ + xx];
}

// ---------------- 1) depthwise 3x3 ----------------
__global__ void dwconv_kernel(const float* __restrict__ vid,
                              const float* __restrict__ qdw,
                              const float* __restrict__ kdw,
                              const float* __restrict__ vdw,
                              float* __restrict__ dwt) {
  int idx = blockIdx.x * 256 + threadIdx.x;           // < 3*3*32*1024
  int branch = idx / (T_ * DIM * H_ * W_);
  int rem = idx % (T_ * DIM * H_ * W_);
  int t = rem / (DIM * H_ * W_);
  int c = (rem >> 10) & 31;
  int y = (rem >> 5) & 31;
  int x = rem & 31;
  const float* w = branch == 0 ? qdw : (branch == 1 ? kdw : vdw);
  float acc = 0.0f;
  #pragma unroll
  for (int ky = 0; ky < 3; ++ky) {
    #pragma unroll
    for (int kx = 0; kx < 3; ++kx) {
      int yy = y + ky - 1, xx = x + kx - 1;
      if ((unsigned)yy < (unsigned)H_ && (unsigned)xx < (unsigned)W_)
        acc += vid[((t * DIM + c) * H_ + yy) * W_ + xx] * w[c * 9 + ky * 3 + kx];
    }
  }
  dwt[idx] = acc;
}

// ---------------- 2) pointwise 1x1 ----------------
__global__ void pwconv_kernel(const float* __restrict__ dwt,
                              const float* __restrict__ qpw, const float* __restrict__ qb,
                              const float* __restrict__ kpw, const float* __restrict__ kb,
                              const float* __restrict__ vpw, const float* __restrict__ vb,
                              float* __restrict__ qf, float* __restrict__ kf,
                              float* __restrict__ vf) {
  int idx = blockIdx.x * 256 + threadIdx.x;
  int branch = idx / (T_ * DIM * H_ * W_);
  int rem = idx % (T_ * DIM * H_ * W_);
  int t = rem / (DIM * H_ * W_);
  int o = (rem >> 10) & 31;
  int pix = rem & 1023;
  const float* pw = branch == 0 ? qpw : (branch == 1 ? kpw : vpw);
  const float* bb = branch == 0 ? qb  : (branch == 1 ? kb  : vb);
  float* outp     = branch == 0 ? qf  : (branch == 1 ? kf  : vf);
  const float* src = dwt + branch * (T_ * DIM * H_ * W_) + t * (DIM * H_ * W_);
  float acc = bb[o];
  #pragma unroll
  for (int c = 0; c < DIM; ++c)
    acc += src[c * 1024 + pix] * pw[o * DIM + c];
  outp[rem] = acc;
}

// ---------------- 3) K-row squared norms ----------------
__global__ void kn_kernel(const float* __restrict__ kfeat, float* __restrict__ kn) {
  int idx = blockIdx.x * 256 + threadIdx.x;            // < 4*3072
  int h = idx / LQ, l = idx % LQ;
  float s = 0.0f;
  for (int pd = 0; pd < PDIM; ++pd) {
    float v = patch_fetch(kfeat, h, l, pd);
    s += v * v;
  }
  kn[idx] = s;
}

// ---------------- 4) pack Q into WMMA A-fragment layout ----------------
// A 16x32 f16 fragment: lane<16 holds row=lane, VGPR j<4 -> K=2j,2j+1 ; j>=4 -> K=2j+8
// lane>=16: same rows shifted by +8 in the low half, +8 more in the high half.
__global__ void packA_kernel(const float* __restrict__ qfeat, _Float16* __restrict__ pA) {
  int idx = blockIdx.x * 256 + threadIdx.x;            // < 4*192*13*512
  int e    = idx & 15;
  int lane = (idx >> 4) & 31;
  int kb   = (idx >> 9) % KBLK;
  int tile = (idx / (512 * KBLK)) % QTILES;
  int h    = idx / (512 * KBLK * QTILES);
  int hi = lane >> 4, rowm = lane & 15;
  int j = e >> 1, rb = e & 1;
  int koff = 2 * j + rb + 8 * hi + (j >= 4 ? 8 : 0);
  int kd = kb * 32 + koff;
  float v = (kd < PDIM) ? patch_fetch(qfeat, h, tile * 16 + rowm, kd) : 0.0f;
  pA[idx] = (_Float16)v;
}

// ---------------- 5) pack K into WMMA B-fragment layout ----------------
// B 32x16 f16 fragment: lane<16 -> col=lane, VGPR j<4 -> K=2j,2j+1 ; j>=4 -> K=2j+8
// lane>=16 -> col=lane-16, same K shifted by +8 (matches A layout).
__global__ void packB_kernel(const float* __restrict__ kfeat, _Float16* __restrict__ pB) {
  int idx = blockIdx.x * 256 + threadIdx.x;
  int e    = idx & 15;
  int lane = (idx >> 4) & 31;
  int kb   = (idx >> 9) % KBLK;
  int tile = (idx / (512 * KBLK)) % QTILES;
  int h    = idx / (512 * KBLK * QTILES);
  int hi = lane >> 4, col = lane & 15;
  int j = e >> 1, rb = e & 1;
  int koff = 2 * j + rb + 8 * hi + (j >= 4 ? 8 : 0);
  int kd = kb * 32 + koff;
  float v = (kd < PDIM) ? patch_fetch(kfeat, h, tile * 16 + col, kd) : 0.0f;
  pB[idx] = (_Float16)v;
}

// ---------------- 6) WMMA GEMM + fused top-K + softmax ----------------
__global__ __launch_bounds__(256)
void gemm_topk_kernel(const _Float16* __restrict__ pA,
                      const _Float16* __restrict__ pB,
                      const float* __restrict__ kn,
                      float* __restrict__ wout, int* __restrict__ iout) {
  __shared__ float S[16 * CHUNK];          // 96 KB score slab (CDNA5: 320KB/WGP)
  __shared__ float cs[16 * 16 * KTOP];
  __shared__ int   ci[16 * 16 * KTOP];

  const int qt   = blockIdx.x;             // 0..191
  const int h    = blockIdx.y;             // 0..3
  const int tid  = threadIdx.x;
  const int lane = tid & 31;
  const int wave = tid >> 5;               // 0..7
  const int hi   = lane >> 4;
  const int nloc = lane & 15;

  // preload all 13 A fragments for this (h, q-tile)
  const _Float16* aBase = pA + ((size_t)(h * QTILES + qt) * KBLK) * 512 + lane * 16;
  v16h a[KBLK];
  #pragma unroll
  for (int kb = 0; kb < KBLK; ++kb)
    a[kb] = *(const v16h*)(aBase + kb * 512);

  const int row = tid >> 4;                // scan-phase row 0..15
  const int sub = tid & 15;

  float ts[KTOP]; int ti[KTOP];
  #pragma unroll
  for (int j = 0; j < KTOP; ++j) { ts[j] = -3.0e38f; ti[j] = 0; }

  for (int ch = 0; ch < NCHUNK; ++ch) {
    // -------- phase 1: WMMA compute of this 16x1536 score chunk --------
    for (int i = 0; i < CTPC / 8; ++i) {
      int ct  = wave + 8 * i;              // col-tile within chunk
      int ctg = ch * CTPC + ct;            // global col-tile
      const _Float16* bBase = pB + ((size_t)(h * QTILES + ctg) * KBLK) * 512 + lane * 16;
      __builtin_prefetch(bBase + KBLK * 8 * 512, 0, 1);   // next tile's B
      v8f acc = {};
      #pragma unroll
      for (int kb = 0; kb < KBLK; ++kb) {
        v16h b = *(const v16h*)(bBase + kb * 512);
        acc = __builtin_amdgcn_wmma_f32_16x16x32_f16(
            false, a[kb], false, b, (short)0, acc, false, false);
      }
      #pragma unroll
      for (int r = 0; r < 8; ++r)
        S[(r + 8 * hi) * CHUNK + ct * 16 + nloc] = acc[r];
    }
    __syncthreads();
    // -------- phase 2: streaming top-K scan (16 threads per row) --------
    for (int colc = sub; colc < CHUNK; colc += 16) {
      int gcol = ch * CHUNK + colc;
      float s = 2.0f * S[row * CHUNK + colc] - kn[h * LQ + gcol];
      int id = gcol;
      if (s > ts[KTOP - 1]) {
        #pragma unroll
        for (int j = 0; j < KTOP; ++j) {
          if (s > ts[j]) {
            float tf = ts[j]; ts[j] = s; s = tf;
            int   tn = ti[j]; ti[j] = id; id = tn;
          }
        }
      }
    }
    __syncthreads();
  }

  // -------- merge 16 partial lists per row, softmax, emit --------
  #pragma unroll
  for (int j = 0; j < KTOP; ++j) {
    cs[(row * 16 + sub) * KTOP + j] = ts[j];
    ci[(row * 16 + sub) * KTOP + j] = ti[j];
  }
  __syncthreads();
  if (tid < 16) {
    int r = tid;
    float best[KTOP]; int bid[KTOP];
    for (int j = 0; j < KTOP; ++j) {
      float bs = -3.0e38f; int bpos = 0;
      for (int m = 0; m < 16 * KTOP; ++m) {
        float v = cs[r * 16 * KTOP + m];
        if (v > bs) { bs = v; bpos = m; }
      }
      bid[j] = ci[r * 16 * KTOP + bpos];
      cs[r * 16 * KTOP + bpos] = -3.0e38f;
      best[j] = bs;
    }
    float mx = best[0], sum = 0.0f, w[KTOP];
    #pragma unroll
    for (int j = 0; j < KTOP; ++j) { w[j] = __expf(best[j] - mx); sum += w[j]; }
    float inv = 1.0f / sum;
    int q = qt * 16 + r;
    size_t base = ((size_t)h * LQ + q) * KTOP;
    #pragma unroll
    for (int j = 0; j < KTOP; ++j) { wout[base + j] = w[j] * inv; iout[base + j] = bid[j]; }
  }
}

// ---------------- 7) weighted V-patch aggregation ----------------
__global__ void agg_kernel(const float* __restrict__ vfeat,
                           const float* __restrict__ wts,
                           const int* __restrict__ inds,
                           float* __restrict__ agg) {
  int blk = blockIdx.x;                    // h*3072 + q
  int h = blk / LQ;
  __shared__ float w[KTOP]; __shared__ int id[KTOP];
  if (threadIdx.x < KTOP) {
    w[threadIdx.x]  = wts[(size_t)blk * KTOP + threadIdx.x];
    id[threadIdx.x] = inds[(size_t)blk * KTOP + threadIdx.x];
  }
  __syncthreads();
  for (int pd = threadIdx.x; pd < PDIM; pd += 128) {
    float acc = 0.0f;
    #pragma unroll
    for (int k = 0; k < KTOP; ++k)
      acc += w[k] * patch_fetch(vfeat, h, id[k], pd);
    agg[(size_t)blk * PDIM + pd] = acc;
  }
}

// ---------------- 8) fold (overlap-add) then projection ----------------
// proj is linear, so fold the 49 patch contributions per channel first,
// then do a single 32x32 projection per pixel.
__global__ void foldproj_kernel(const float* __restrict__ agg,
                                const float* __restrict__ projw,
                                const float* __restrict__ projb,
                                float* __restrict__ out) {
  int q = blockIdx.x;                      // pixel index
  int t = q >> 10, y = (q >> 5) & 31, x = q & 31;
  int d = threadIdx.x;                     // channel 0..31
  int h = d >> 3, ed = d & 7;
  __shared__ float fl[DIM];
  float acc = 0.0f; int cnt = 0;
  for (int di = 0; di < PS; ++di) {
    for (int dj = 0; dj < PS; ++dj) {
      int yy = y + PAD_ - di, xx = x + PAD_ - dj;
      if ((unsigned)yy < (unsigned)H_ && (unsigned)xx < (unsigned)W_) {
        int qp = t * 1024 + yy * 32 + xx;
        acc += agg[((size_t)h * LQ + qp) * PDIM + ed * PS2 + di * PS + dj];
        ++cnt;
      }
    }
  }
  fl[d] = acc;
  __syncthreads();
  float o = 0.0f;
  #pragma unroll
  for (int dd = 0; dd < DIM; ++dd) o += projw[d * DIM + dd] * fl[dd];
  o = o / (float)cnt + projb[d];
  out[((t * DIM + d) * H_ + y) * W_ + x] = o;
}

// ---------------- launch ----------------
extern "C" void kernel_launch(void* const* d_in, const int* in_sizes, int n_in,
                              void* d_out, int out_size, void* d_ws, size_t ws_size,
                              hipStream_t stream) {
  (void)in_sizes; (void)n_in; (void)out_size; (void)ws_size;
  const float* vid   = (const float*)d_in[0];
  const float* q_dw  = (const float*)d_in[1];
  const float* q_pw  = (const float*)d_in[2];
  const float* q_b   = (const float*)d_in[3];
  const float* k_dw  = (const float*)d_in[4];
  const float* k_pw  = (const float*)d_in[5];
  const float* k_b   = (const float*)d_in[6];
  const float* v_dw  = (const float*)d_in[7];
  const float* v_pw  = (const float*)d_in[8];
  const float* v_b   = (const float*)d_in[9];
  const float* pr_w  = (const float*)d_in[10];
  const float* pr_b  = (const float*)d_in[11];
  float* out = (float*)d_out;

  char* ws = (char*)d_ws;
  float*    qfeat = (float*)(ws + 0);
  float*    kfeat = (float*)(ws + 393216);
  float*    vfeat = (float*)(ws + 786432);
  float*    dwt   = (float*)(ws + 1179648);
  _Float16* packA = (_Float16*)(ws + 2359296);
  _Float16* packB = (_Float16*)(ws + 12582912);
  float*    kn    = (float*)(ws + 22806528);
  float*    wts   = (float*)(ws + 22855680);
  int*      inds  = (int*)  (ws + 23347200);
  float*    agg   = (float*)(ws + 23838720);   // end: ~43.1 MB

  dwconv_kernel<<<1152, 256, 0, stream>>>(vid, q_dw, k_dw, v_dw, dwt);
  pwconv_kernel<<<1152, 256, 0, stream>>>(dwt, q_pw, q_b, k_pw, k_b, v_pw, v_b,
                                          qfeat, kfeat, vfeat);
  kn_kernel<<<48, 256, 0, stream>>>(kfeat, kn);
  packA_kernel<<<19968, 256, 0, stream>>>(qfeat, packA);
  packB_kernel<<<19968, 256, 0, stream>>>(kfeat, packB);
  gemm_topk_kernel<<<dim3(QTILES, NHEADS), 256, 0, stream>>>(packA, packB, kn, wts, inds);
  agg_kernel<<<NHEADS * LQ, 128, 0, stream>>>(vfeat, wts, inds, agg);
  foldproj_kernel<<<LQ, DIM, 0, stream>>>(agg, pr_w, pr_b, out);
}
